// CostToGoHead_38886633898508
// MI455X (gfx1250) — matrix operand order repaired and checked by reading, not compiled, not run on hardware
//
#include <hip/hip_runtime.h>
#include <hip/hip_bf16.h>

typedef __attribute__((ext_vector_type(16))) _Float16 v16h;
typedef __attribute__((ext_vector_type(8)))  _Float16 h8;
typedef __attribute__((ext_vector_type(8)))  float    v8f;
typedef __attribute__((ext_vector_type(2)))  __fp16   v2hf;   // cvt_pkrtz result type

#define N_NODES 1024
#define D_IN    128
#define R_DIM   64
#define H1_DIM  128   // 2R
#define H2_DIM  64    // R
#define EPSV    1e-5f

// ---------------------------------------------------------------------------
// Phase 0: convert weight slices to f16 for WMMA fragment loads
//   w1c_h[m][k] = w1[m, 128+k]   (m=0..127, k=0..63)   A of GEMM1 (h1^T form)
//   w2_h [q][m] = w2[q, m]       (q=0..63,  m=0..127)  A of GEMM2 (h2^T form)
// ---------------------------------------------------------------------------
__global__ void prep_weights_kernel(const float* __restrict__ w1,
                                    const float* __restrict__ w2,
                                    _Float16* __restrict__ w1c_h,
                                    _Float16* __restrict__ w2_h) {
  int tid = threadIdx.x;
  for (int idx = tid; idx < H1_DIM * R_DIM; idx += blockDim.x) {
    int m = idx >> 6, k = idx & 63;
    w1c_h[idx] = (_Float16)w1[m * 192 + 128 + k];
  }
  for (int idx = tid; idx < H2_DIM * H1_DIM; idx += blockDim.x) {
    w2_h[idx] = (_Float16)w2[idx];
  }
}

// ---------------------------------------------------------------------------
// Phase 1: per-node projections + layernorm + relu (f32), f16 copies, and
// factored per-node parts of MLP layer 1:
//   Afull[i][m] = b1[m] + sum_k src[i,k]*w1[m,k]
//   Bfull[j][m] =         sum_k dst[j,k]*w1[m,64+k]
// One block of 128 threads per node row.
// ---------------------------------------------------------------------------
__global__ void node_proj_kernel(const float* __restrict__ node_emb,
    const float* __restrict__ w_src, const float* __restrict__ b_src,
    const float* __restrict__ g_src, const float* __restrict__ be_src,
    const float* __restrict__ w_dst, const float* __restrict__ b_dst,
    const float* __restrict__ g_dst, const float* __restrict__ be_dst,
    const float* __restrict__ w1, const float* __restrict__ b1,
    _Float16* __restrict__ srch, _Float16* __restrict__ dsth,
    float* __restrict__ Afull, float* __restrict__ Bfull) {
  __shared__ float sx[D_IN];
  __shared__ float sv[2][R_DIM];
  const int row = blockIdx.x;
  const int tid = threadIdx.x;              // 0..127
  sx[tid] = node_emb[row * D_IN + tid];
  __syncthreads();

  const int half = tid >> 6;                // 0 = src, 1 = dst
  const int o    = tid & 63;
  const float* W = half ? w_dst : w_src;
  const float* B = half ? b_dst : b_src;
  float acc = B[o];
  #pragma unroll 4
  for (int k = 0; k < D_IN; ++k) acc += sx[k] * W[o * D_IN + k];
  sv[half][o] = acc;
  __syncthreads();

  float mean = 0.f;
  for (int k = 0; k < R_DIM; ++k) mean += sv[half][k];
  mean *= (1.0f / R_DIM);
  float var = 0.f;
  for (int k = 0; k < R_DIM; ++k) { float d = sv[half][k] - mean; var += d * d; }
  var *= (1.0f / R_DIM);
  const float* G  = half ? g_dst : g_src;
  const float* BE = half ? be_dst : be_src;
  float val = (acc - mean) * rsqrtf(var + EPSV) * G[o] + BE[o];
  val = fmaxf(val, 0.f);
  __syncthreads();
  sv[half][o] = val;
  if (half) dsth[row * R_DIM + o] = (_Float16)val;
  else      srch[row * R_DIM + o] = (_Float16)val;
  __syncthreads();

  float a    = b1[tid];
  float bsum = 0.f;
  #pragma unroll 4
  for (int k = 0; k < R_DIM; ++k) {
    a    += sv[0][k] * w1[tid * 192 + k];
    bsum += sv[1][k] * w1[tid * 192 + 64 + k];
  }
  Afull[row * H1_DIM + tid] = a;
  Bfull[row * H1_DIM + tid] = bsum;
}

// ---------------------------------------------------------------------------
// Phase 2: pairwise MLP, transposed form (outputs on M, pairs on N), no LDS.
// Block = 256 threads (8 waves). Wave owns 16 consecutive j; loops 16 i's.
// Per 16-pair task:
//   GEMM1: h1^T[128 x 16] = relu(W1c[128x64] x P^T[64x16] + A_i + B_j)
//   GEMM2: h2^T[64  x 16] = relu(w2 [64x128] x h1^T       + b2)
//   GEMM3: cost[16] = w3 . h2  (VALU fma + single cross-half shfl)
// h1^T D-layout -> GEMM2 B-layout needs only a lane^16 exchange (no LDS).
// ---------------------------------------------------------------------------
__global__ __launch_bounds__(256) void pair_mlp_kernel(
    const _Float16* __restrict__ srch, const _Float16* __restrict__ dsth,
    const float* __restrict__ Afull,  const float* __restrict__ Bfull,
    const _Float16* __restrict__ w1c_h, const _Float16* __restrict__ w2_h,
    const float* __restrict__ b2, const float* __restrict__ w3,
    const float* __restrict__ b3, float* __restrict__ out) {
  const int wave = threadIdx.x >> 5;
  const int lane = threadIdx.x & 31;
  const int g    = lane >> 4;               // lane group (ISA layouts)
  const int hn   = lane & 15;
  const int j0   = blockIdx.x * 128 + wave * 16;
  const int i0   = blockIdx.y * 16;

  // ---- per-wave invariants (compiler hoists into VGPRs) ----
  // dst chunks for P^T B-fragments: lane = pair column p = hn -> j = j0+hn,
  // halves k = 32s + 16g + (0..15), contiguous.
  const _Float16* dp = dsth + (j0 + hn) * R_DIM;
  h8 dlo0 = *(const h8*)(dp      + 16 * g);
  h8 dhi0 = *(const h8*)(dp + 8  + 16 * g);
  h8 dlo1 = *(const h8*)(dp + 32 + 16 * g);
  h8 dhi1 = *(const h8*)(dp + 40 + 16 * g);
  // Bfull row for this lane's pair (i-invariant): m = 16t + 8g + v
  v8f bfv[8];
  #pragma unroll
  for (int t = 0; t < 8; ++t)
    bfv[t] = *(const v8f*)(Bfull + (j0 + hn) * H1_DIM + 16 * t + 8 * g);
  // b2 / w3 slices for transposed C/D layout: q = 16t2 + 8g + v
  v8f bb2[4], w3v[4];
  #pragma unroll
  for (int t = 0; t < 4; ++t) {
    bb2[t] = *(const v8f*)(b2 + 16 * t + 8 * g);
    w3v[t] = *(const v8f*)(w3 + 16 * t + 8 * g);
  }
  const float b3v = b3[0];

  for (int ii = 0; ii < 16; ++ii) {
    const int i = i0 + ii;

    // ---- P^T B-fragments: pfrag[s], halves k = 32s + 16g + h ----
    const _Float16* sp = srch + i * R_DIM;
    h8 slo0 = *(const h8*)(sp      + 16 * g);
    h8 shi0 = *(const h8*)(sp + 8  + 16 * g);
    h8 slo1 = *(const h8*)(sp + 32 + 16 * g);
    h8 shi1 = *(const h8*)(sp + 40 + 16 * g);
    h8 pl0 = slo0 * dlo0, ph0 = shi0 * dhi0;
    h8 pl1 = slo1 * dlo1, ph1 = shi1 * dhi1;
    v16h pfrag[2];
    pfrag[0] = __builtin_shufflevector(pl0, ph0, 0,1,2,3,4,5,6,7,8,9,10,11,12,13,14,15);
    pfrag[1] = __builtin_shufflevector(pl1, ph1, 0,1,2,3,4,5,6,7,8,9,10,11,12,13,14,15);

    // ---- acc1 init: D-layout row m = 16t + 8g + v, col pair = hn ----
    v8f acc1[8];
    #pragma unroll
    for (int t = 0; t < 8; ++t) {
      v8f afv = *(const v8f*)(Afull + i * H1_DIM + 16 * t + 8 * g);
      acc1[t] = afv + bfv[t];
    }
    // ---- GEMM1: A = W1c fragments (A-layout: row m = 16t+hn,
    //      k chunks at 32s+16g and 32s+16g+8), B = pfrag[s] ----
    #pragma unroll
    for (int s = 0; s < 2; ++s) {
      #pragma unroll
      for (int t = 0; t < 8; ++t) {
        const _Float16* wp = w1c_h + (16 * t + hn) * R_DIM + 32 * s + 16 * g;
        h8 alo = *(const h8*)wp;
        h8 ahi = *(const h8*)(wp + 8);
        v16h afr = __builtin_shufflevector(alo, ahi, 0,1,2,3,4,5,6,7,8,9,10,11,12,13,14,15);
        acc1[t] = __builtin_amdgcn_wmma_f32_16x16x32_f16(
            false, afr, false, pfrag[s], (short)0, acc1[t], false, false);
      }
    }

    // ---- relu + pack h1^T to f16 dwords: pkd[t][d] = (m=16t+8g+2d, +1) ----
    unsigned pkd[8][4];
    #pragma unroll
    for (int t = 0; t < 8; ++t)
      #pragma unroll
      for (int d = 0; d < 4; ++d) {
        v2hf p = __builtin_amdgcn_cvt_pkrtz(fmaxf(acc1[t][2 * d], 0.f),
                                            fmaxf(acc1[t][2 * d + 1], 0.f));
        pkd[t][d] = __builtin_bit_cast(unsigned, p);
      }

    // ---- GEMM2: A = w2 fragments, B = h1^T via lane^16 exchange ----
    // B2 frag s2 needs halves m = 32*s2 + 16g + (0..15); that whole range is
    // m-tile t* = 2*s2 + g: low 8 halves live in g=0 lanes, high 8 in g=1.
    v8f acc2[4];
    #pragma unroll
    for (int s2 = 0; s2 < 4; ++s2) {
      unsigned low[4], high[4];
      #pragma unroll
      for (int d = 0; d < 4; ++d) {
        unsigned A = pkd[2 * s2][d];          // tile 2*s2 (this lane's block)
        unsigned Bk = pkd[2 * s2 + 1][d];     // tile 2*s2+1
        unsigned snd = g ? A : Bk;            // what the partner lane needs
        unsigned rcv = (unsigned)__shfl_xor((int)snd, 16, 32);
        low[d]  = g ? rcv : A;                // m = 32s2+16g   .. +7
        high[d] = g ? Bk  : rcv;              // m = 32s2+16g+8 .. +15
      }
      union { unsigned u[8]; v16h v; } bu;
      #pragma unroll
      for (int d = 0; d < 4; ++d) { bu.u[d] = low[d]; bu.u[4 + d] = high[d]; }

      #pragma unroll
      for (int t2 = 0; t2 < 4; ++t2) {
        const _Float16* wp = w2_h + (16 * t2 + hn) * H1_DIM + 32 * s2 + 16 * g;
        h8 alo = *(const h8*)wp;
        h8 ahi = *(const h8*)(wp + 8);
        v16h afr = __builtin_shufflevector(alo, ahi, 0,1,2,3,4,5,6,7,8,9,10,11,12,13,14,15);
        v8f cin = (s2 == 0) ? bb2[t2] : acc2[t2];
        acc2[t2] = __builtin_amdgcn_wmma_f32_16x16x32_f16(
            false, afr, false, bu.v, (short)0, cin, false, false);
      }
    }

    // ---- GEMM3: cost[pair=hn] = b3 + sum_q relu(h2[q,pair]) * w3[q] ----
    float partial = 0.f;
    #pragma unroll
    for (int t2 = 0; t2 < 4; ++t2)
      #pragma unroll
      for (int v = 0; v < 8; ++v)
        partial += fmaxf(acc2[t2][v], 0.f) * w3v[t2][v];
    float tot = partial + __shfl_xor(partial, 16, 32);   // join the two halves
    if (g == 0) out[i * N_NODES + j0 + hn] = tot + b3v;  // coalesced 16-lane row
  }
}

// ---------------------------------------------------------------------------
extern "C" void kernel_launch(void* const* d_in, const int* in_sizes, int n_in,
                              void* d_out, int out_size, void* d_ws, size_t ws_size,
                              hipStream_t stream) {
  const float* node_emb = (const float*)d_in[0];
  const float* w_src  = (const float*)d_in[1];
  const float* b_src  = (const float*)d_in[2];
  const float* g_src  = (const float*)d_in[3];
  const float* be_src = (const float*)d_in[4];
  const float* w_dst  = (const float*)d_in[5];
  const float* b_dst  = (const float*)d_in[6];
  const float* g_dst  = (const float*)d_in[7];
  const float* be_dst = (const float*)d_in[8];
  const float* w1 = (const float*)d_in[9];
  const float* b1 = (const float*)d_in[10];
  const float* w2 = (const float*)d_in[11];
  const float* b2 = (const float*)d_in[12];
  const float* w3 = (const float*)d_in[13];
  const float* b3 = (const float*)d_in[14];

  char* ws = (char*)d_ws;
  _Float16* srch  = (_Float16*)(ws + 0);        // 1024*64*2  = 128 KB
  _Float16* dsth  = (_Float16*)(ws + 131072);   // 128 KB
  _Float16* w1c_h = (_Float16*)(ws + 262144);   // 16 KB
  _Float16* w2_h  = (_Float16*)(ws + 278528);   // 16 KB
  float*    Afull = (float*)  (ws + 294912);    // 1024*128*4 = 512 KB
  float*    Bfull = (float*)  (ws + 819200);    // 512 KB
  float*    out   = (float*)d_out;

  prep_weights_kernel<<<dim3(1), dim3(256), 0, stream>>>(w1, w2, w1c_h, w2_h);
  node_proj_kernel<<<dim3(N_NODES), dim3(128), 0, stream>>>(
      node_emb, w_src, b_src, g_src, be_src, w_dst, b_dst, g_dst, be_dst,
      w1, b1, srch, dsth, Afull, Bfull);
  pair_mlp_kernel<<<dim3(8, 64), dim3(256), 0, stream>>>(
      srch, dsth, Afull, Bfull, w1c_h, w2_h, b2, w3, b3, out);
}
